// MyNN_22557168239387
// MI455X (gfx1250) — matrix operand order repaired and checked by reading, not compiled, not run on hardware
//
#include <hip/hip_runtime.h>
#include <hip/hip_bf16.h>

typedef __attribute__((ext_vector_type(16))) _Float16 v16h;
typedef __attribute__((ext_vector_type(8)))  _Float16 v8h;
typedef __attribute__((ext_vector_type(8)))  float    v8f;

#define WAVES 8          // 256 threads = 8 wave32
#define ROWS_PER_BLOCK 128

// ---------------------------------------------------------------------------
// WMMA helpers (CDNA5 16x16x32 f16 -> f32)
// ---------------------------------------------------------------------------
__device__ __forceinline__ v8f wmma32(v16h a, v16h b, v8f c) {
    return __builtin_amdgcn_wmma_f32_16x16x32_f16(
        /*neg_a=*/false, a, /*neg_b=*/false, b,
        /*c_mod=*/(short)0, c, /*reuse_a=*/false, /*reuse_b=*/false);
}

// A fragment (16xK tile, row-major f16 in LDS, row stride ldk halves).
// Layout: lanes 0-15 hold M=lane, K = kc+0..15;
// lanes 16-31 hold M=lane-16, K = kc+16..31.
__device__ __forceinline__ v16h load_A_lds(const _Float16* base, int lane,
                                           int kchunk, int ldk) {
    const _Float16* p = base + (lane & 15) * ldk + kchunk + ((lane >> 4) << 4);
    return *(const v16h*)p;          // 32B aligned contiguous K run
}

// B fragment (KxN tile), weights stored transposed [N][Kpad] f16 in LDS.
// lanes 0-15: N=lane, K = kc+0..15 ; lanes 16-31: N=lane-16, K = kc+16..31.
__device__ __forceinline__ v16h load_B_lds(const _Float16* w, int lane,
                                           int ntile, int kchunk, int ldk) {
    const _Float16* p = w + (((ntile << 4) + (lane & 15)) * ldk)
                          + kchunk + ((lane >> 4) << 4);
    return *(const v16h*)p;          // 32B aligned contiguous K run
}

// ---------------------------------------------------------------------------
// Weight convert + transpose: W[K][N] f32 -> Wt[N][Kpad] f16 (zero-padded K)
// ---------------------------------------------------------------------------
__global__ void convertT_kernel(const float* __restrict__ W, _Float16* __restrict__ Wt,
                                int K, int N, int Kpad) {
    int idx = blockIdx.x * blockDim.x + threadIdx.x;
    if (idx >= N * Kpad) return;
    int n = idx / Kpad, k = idx % Kpad;
    Wt[idx] = (k < K) ? (_Float16)W[k * N + n] : (_Float16)0.f;
}

// ---------------------------------------------------------------------------
// Degrees via float atomics
// ---------------------------------------------------------------------------
__global__ void degree_kernel(const int* __restrict__ src, const int* __restrict__ dst,
                              float* __restrict__ dego, float* __restrict__ degi, int E) {
    int e = blockIdx.x * blockDim.x + threadIdx.x;
    if (e >= E) return;
    atomicAdd(&dego[src[e]], 1.0f);
    atomicAdd(&degi[dst[e]], 1.0f);
}

// ---------------------------------------------------------------------------
// Node MLP: x[M,32] -> relu -> relu -> hn[M,64].  wt = {w0t(64x32), w1t(64x64), w2t(64x64)}
// ---------------------------------------------------------------------------
__global__ void __launch_bounds__(256)
node_mlp_kernel(const float* __restrict__ x, const _Float16* __restrict__ wt,
                const float* __restrict__ b0, const float* __restrict__ b1,
                const float* __restrict__ b2, float* __restrict__ hn, int M) {
    __shared__ _Float16 s_w[10240];
    __shared__ _Float16 s_in[WAVES][16 * 32];
    __shared__ _Float16 s_h [WAVES][16 * 64];
    __shared__ _Float16 s_h2[WAVES][16 * 64];
    int tid = threadIdx.x;
    for (int i = tid; i < 10240; i += 256) s_w[i] = wt[i];
    __syncthreads();

    int wave = tid >> 5, lane = tid & 31;
    int mbase = (blockIdx.x * WAVES + wave) * 16;

    _Float16* in = s_in[wave];
    for (int idx = lane; idx < 16 * 32; idx += 32) {
        int r = idx >> 5, k = idx & 31;
        int row = mbase + r;
        in[idx] = (_Float16)((row < M) ? x[(size_t)row * 32 + k] : 0.f);
    }

    // layer 0: K = 32 (one chunk)
    v8f c0[4] = {};
    {
        v16h a = load_A_lds(in, lane, 0, 32);
#pragma unroll
        for (int n = 0; n < 4; ++n)
            c0[n] = wmma32(a, load_B_lds(s_w, lane, n, 0, 32), c0[n]);
    }
    _Float16* h = s_h[wave];
#pragma unroll
    for (int n = 0; n < 4; ++n) {
        int col = (n << 4) + (lane & 15);
        float bb = b0[col];
#pragma unroll
        for (int r = 0; r < 8; ++r) {
            int m = r + ((lane >> 4) << 3);
            float v = c0[n][r] + bb;
            h[m * 64 + col] = (_Float16)(v > 0.f ? v : 0.f);
        }
    }

    // layer 1: K = 64 (two chunks)
    v8f c1[4] = {};
#pragma unroll
    for (int kc = 0; kc < 64; kc += 32) {
        v16h a = load_A_lds(h, lane, kc, 64);
#pragma unroll
        for (int n = 0; n < 4; ++n)
            c1[n] = wmma32(a, load_B_lds(s_w + 2048, lane, n, kc, 64), c1[n]);
    }
    _Float16* h2 = s_h2[wave];
#pragma unroll
    for (int n = 0; n < 4; ++n) {
        int col = (n << 4) + (lane & 15);
        float bb = b1[col];
#pragma unroll
        for (int r = 0; r < 8; ++r) {
            int m = r + ((lane >> 4) << 3);
            float v = c1[n][r] + bb;
            h2[m * 64 + col] = (_Float16)(v > 0.f ? v : 0.f);
        }
    }

    // layer 2: K = 64, no relu, write f32
    v8f c2[4] = {};
#pragma unroll
    for (int kc = 0; kc < 64; kc += 32) {
        v16h a = load_A_lds(h2, lane, kc, 64);
#pragma unroll
        for (int n = 0; n < 4; ++n)
            c2[n] = wmma32(a, load_B_lds(s_w + 6144, lane, n, kc, 64), c2[n]);
    }
#pragma unroll
    for (int n = 0; n < 4; ++n) {
        int col = (n << 4) + (lane & 15);
        float bb = b2[col];
#pragma unroll
        for (int r = 0; r < 8; ++r) {
            int m = r + ((lane >> 4) << 3);
            int row = mbase + m;
            if (row < M) hn[(size_t)row * 64 + col] = c2[n][r] + bb;
        }
    }
}

// ---------------------------------------------------------------------------
// Fused edge MLP + conv1 message scatter.
// he tile stays in WMMA accumulators; msg = (hn[s]+hn[d]+he)*norm -> atomic agg1[d].
// ---------------------------------------------------------------------------
__global__ void __launch_bounds__(256)
edge_fused_kernel(const float* __restrict__ efeats, int Kin, int Ecount, int eoff,
                  const _Float16* __restrict__ wt,
                  const float* __restrict__ b0, const float* __restrict__ b1,
                  const float* __restrict__ b2,
                  const float* __restrict__ hn,
                  const int* __restrict__ src, const int* __restrict__ dst,
                  const float* __restrict__ dego, const float* __restrict__ degi,
                  float* __restrict__ agg1) {
    __shared__ _Float16 s_w[10240];
    __shared__ _Float16 s_in[WAVES][16 * 32];
    __shared__ _Float16 s_h [WAVES][16 * 64];
    __shared__ _Float16 s_h2[WAVES][16 * 64];
    int tid = threadIdx.x;
    for (int i = tid; i < 10240; i += 256) s_w[i] = wt[i];
    __syncthreads();

    int wave = tid >> 5, lane = tid & 31;
    int ebase = (blockIdx.x * WAVES + wave) * 16;

    _Float16* in = s_in[wave];
    for (int idx = lane; idx < 16 * 32; idx += 32) {
        int r = idx >> 5, k = idx & 31;
        int e = ebase + r;
        float v = (e < Ecount && k < Kin) ? efeats[(size_t)e * Kin + k] : 0.f;
        in[idx] = (_Float16)v;
    }

    v8f c0[4] = {};
    {
        v16h a = load_A_lds(in, lane, 0, 32);
#pragma unroll
        for (int n = 0; n < 4; ++n)
            c0[n] = wmma32(a, load_B_lds(s_w, lane, n, 0, 32), c0[n]);
    }
    _Float16* h = s_h[wave];
#pragma unroll
    for (int n = 0; n < 4; ++n) {
        int col = (n << 4) + (lane & 15);
        float bb = b0[col];
#pragma unroll
        for (int r = 0; r < 8; ++r) {
            int m = r + ((lane >> 4) << 3);
            float v = c0[n][r] + bb;
            h[m * 64 + col] = (_Float16)(v > 0.f ? v : 0.f);
        }
    }

    v8f c1[4] = {};
#pragma unroll
    for (int kc = 0; kc < 64; kc += 32) {
        v16h a = load_A_lds(h, lane, kc, 64);
#pragma unroll
        for (int n = 0; n < 4; ++n)
            c1[n] = wmma32(a, load_B_lds(s_w + 2048, lane, n, kc, 64), c1[n]);
    }
    _Float16* h2 = s_h2[wave];
#pragma unroll
    for (int n = 0; n < 4; ++n) {
        int col = (n << 4) + (lane & 15);
        float bb = b1[col];
#pragma unroll
        for (int r = 0; r < 8; ++r) {
            int m = r + ((lane >> 4) << 3);
            float v = c1[n][r] + bb;
            h2[m * 64 + col] = (_Float16)(v > 0.f ? v : 0.f);
        }
    }

    v8f c2[4] = {};
#pragma unroll
    for (int kc = 0; kc < 64; kc += 32) {
        v16h a = load_A_lds(h2, lane, kc, 64);
#pragma unroll
        for (int n = 0; n < 4; ++n)
            c2[n] = wmma32(a, load_B_lds(s_w + 6144, lane, n, kc, 64), c2[n]);
    }

    // scatter epilogue: msg = (hn[src]+hn[dst]+he) * rsqrt(deg_out*deg_in)
    int colbase = lane & 15;
    float bb[4];
#pragma unroll
    for (int n = 0; n < 4; ++n) bb[n] = b2[(n << 4) + colbase];
#pragma unroll
    for (int r = 0; r < 8; ++r) {
        int m = r + ((lane >> 4) << 3);
        int e = ebase + m;
        if (e < Ecount) {
            int eg = e + eoff;
            int s = src[eg], d = dst[eg];
            float go = dego[s]; go = go > 1.f ? go : 1.f;
            float gi = degi[d]; gi = gi > 1.f ? gi : 1.f;
            float ne = rsqrtf(go * gi);
#pragma unroll
            for (int n = 0; n < 4; ++n) {
                int col = (n << 4) + colbase;
                float val = (hn[(size_t)s * 64 + col] + hn[(size_t)d * 64 + col]
                             + c2[n][r] + bb[n]) * ne;
                atomicAdd(&agg1[(size_t)d * 64 + col], val);
            }
        }
    }
}

// ---------------------------------------------------------------------------
// conv1 GEMM: h1 = relu(agg1 @ c1_w + c1_b)   [M,64]x[64,64]
// ---------------------------------------------------------------------------
__global__ void __launch_bounds__(256)
conv1_kernel(const float* __restrict__ agg1, const _Float16* __restrict__ wt,
             const float* __restrict__ bias, float* __restrict__ h1, int M) {
    __shared__ _Float16 s_w[4096];
    __shared__ _Float16 s_in[WAVES][16 * 64];
    int tid = threadIdx.x;
    for (int i = tid; i < 4096; i += 256) s_w[i] = wt[i];
    __syncthreads();

    int wave = tid >> 5, lane = tid & 31;
    int mbase = (blockIdx.x * WAVES + wave) * 16;

    _Float16* in = s_in[wave];
    for (int idx = lane; idx < 16 * 64; idx += 32) {
        int r = idx >> 6, k = idx & 63;
        int row = mbase + r;
        in[idx] = (_Float16)((row < M) ? agg1[(size_t)row * 64 + k] : 0.f);
    }

    v8f c[4] = {};
#pragma unroll
    for (int kc = 0; kc < 64; kc += 32) {
        v16h a = load_A_lds(in, lane, kc, 64);
#pragma unroll
        for (int n = 0; n < 4; ++n)
            c[n] = wmma32(a, load_B_lds(s_w, lane, n, kc, 64), c[n]);
    }
#pragma unroll
    for (int n = 0; n < 4; ++n) {
        int col = (n << 4) + (lane & 15);
        float bb = bias[col];
#pragma unroll
        for (int r = 0; r < 8; ++r) {
            int m = r + ((lane >> 4) << 3);
            int row = mbase + m;
            if (row < M) {
                float v = c[n][r] + bb;
                h1[(size_t)row * 64 + col] = v > 0.f ? v : 0.f;
            }
        }
    }
}

// ---------------------------------------------------------------------------
// conv2 scatter: agg2[dst] += h1[src] * norm_e  (one thread per edge-element)
// ---------------------------------------------------------------------------
__global__ void conv2_scatter_kernel(const float* __restrict__ h1,
                                     const int* __restrict__ src, const int* __restrict__ dst,
                                     const float* __restrict__ dego, const float* __restrict__ degi,
                                     float* __restrict__ agg2, int E) {
    long long idx = (long long)blockIdx.x * blockDim.x + threadIdx.x;
    if (idx >= (long long)E * 64) return;
    int e = (int)(idx >> 6), col = (int)(idx & 63);
    int s = src[e], d = dst[e];
    float go = dego[s]; go = go > 1.f ? go : 1.f;
    float gi = degi[d]; gi = gi > 1.f ? gi : 1.f;
    float ne = rsqrtf(go * gi);
    atomicAdd(&agg2[(size_t)d * 64 + col], h1[(size_t)s * 64 + col] * ne);
}

// ---------------------------------------------------------------------------
// final: out = agg2 @ c2_w + c2_b   [M,64]x[64,1]
// ---------------------------------------------------------------------------
__global__ void final_kernel(const float* __restrict__ agg2, const float* __restrict__ w,
                             const float* __restrict__ b, float* __restrict__ out, int M) {
    int i = blockIdx.x * blockDim.x + threadIdx.x;
    if (i >= M) return;
    const float4* a  = (const float4*)(agg2 + (size_t)i * 64);
    const float4* wv = (const float4*)w;
    float sum = 0.f;
#pragma unroll
    for (int j = 0; j < 16; ++j) {
        float4 av = a[j], w4 = wv[j];
        sum += av.x * w4.x + av.y * w4.y + av.z * w4.z + av.w * w4.w;
    }
    out[i] = sum + b[0];
}

// ---------------------------------------------------------------------------
extern "C" void kernel_launch(void* const* d_in, const int* in_sizes, int n_in,
                              void* d_out, int out_size, void* d_ws, size_t ws_size,
                              hipStream_t stream) {
    const float* node_feats = (const float*)d_in[0];
    const float* ef0  = (const float*)d_in[1];
    const float* ef1  = (const float*)d_in[2];
    const float* n_w0 = (const float*)d_in[3];  const float* n_b0 = (const float*)d_in[4];
    const float* n_w1 = (const float*)d_in[5];  const float* n_b1 = (const float*)d_in[6];
    const float* n_w2 = (const float*)d_in[7];  const float* n_b2 = (const float*)d_in[8];
    const float* e0_w0 = (const float*)d_in[9];  const float* e0_b0 = (const float*)d_in[10];
    const float* e0_w1 = (const float*)d_in[11]; const float* e0_b1 = (const float*)d_in[12];
    const float* e0_w2 = (const float*)d_in[13]; const float* e0_b2 = (const float*)d_in[14];
    const float* e1_w0 = (const float*)d_in[15]; const float* e1_b0 = (const float*)d_in[16];
    const float* e1_w1 = (const float*)d_in[17]; const float* e1_b1 = (const float*)d_in[18];
    const float* e1_w2 = (const float*)d_in[19]; const float* e1_b2 = (const float*)d_in[20];
    const float* c1_w = (const float*)d_in[21];  const float* c1_b = (const float*)d_in[22];
    const float* c2_w = (const float*)d_in[23];  const float* c2_b = (const float*)d_in[24];
    const int*   src  = (const int*)d_in[25];
    const int*   dst  = (const int*)d_in[26];
    float* out = (float*)d_out;

    const int N = 50000, E0 = 500000, E1 = 300000, E = E0 + E1, D = 64;

    // workspace layout
    char* ws = (char*)d_ws;
    _Float16* wt = (_Float16*)ws;                         // 34816 halves (<128KB)
    float* hn   = (float*)(ws + 131072);                  // N*D
    float* agg1 = hn   + (size_t)N * D;                   // N*D
    float* h1   = agg1 + (size_t)N * D;                   // N*D
    float* agg2 = h1   + (size_t)N * D;                   // N*D
    float* dego = agg2 + (size_t)N * D;                   // N
    float* degi = dego + N;                               // N

    hipMemsetAsync(agg1, 0, (size_t)N * D * sizeof(float), stream);
    hipMemsetAsync(agg2, 0, (size_t)N * D * sizeof(float), stream);
    hipMemsetAsync(dego, 0, (size_t)N * 2 * sizeof(float), stream);

    // convert+transpose weights to f16 [N][Kpad]
    auto convT = [&](const float* W, _Float16* Wt, int K, int Nn, int Kpad) {
        int total = Nn * Kpad;
        convertT_kernel<<<(total + 255) / 256, 256, 0, stream>>>(W, Wt, K, Nn, Kpad);
    };
    convT(n_w0,  wt + 0,     32, 64, 32);
    convT(n_w1,  wt + 2048,  64, 64, 64);
    convT(n_w2,  wt + 6144,  64, 64, 64);
    convT(e0_w0, wt + 10240, 16, 64, 32);
    convT(e0_w1, wt + 12288, 64, 64, 64);
    convT(e0_w2, wt + 16384, 64, 64, 64);
    convT(e1_w0, wt + 20480,  8, 64, 32);
    convT(e1_w1, wt + 22528, 64, 64, 64);
    convT(e1_w2, wt + 26624, 64, 64, 64);
    convT(c1_w,  wt + 30720, 64, 64, 64);

    degree_kernel<<<(E + 255) / 256, 256, 0, stream>>>(src, dst, dego, degi, E);

    node_mlp_kernel<<<(N + ROWS_PER_BLOCK - 1) / ROWS_PER_BLOCK, 256, 0, stream>>>(
        node_feats, wt, n_b0, n_b1, n_b2, hn, N);

    edge_fused_kernel<<<(E0 + ROWS_PER_BLOCK - 1) / ROWS_PER_BLOCK, 256, 0, stream>>>(
        ef0, 16, E0, 0, wt + 10240, e0_b0, e0_b1, e0_b2,
        hn, src, dst, dego, degi, agg1);
    edge_fused_kernel<<<(E1 + ROWS_PER_BLOCK - 1) / ROWS_PER_BLOCK, 256, 0, stream>>>(
        ef1, 8, E1, E0, wt + 20480, e1_b0, e1_b1, e1_b2,
        hn, src, dst, dego, degi, agg1);

    conv1_kernel<<<(N + ROWS_PER_BLOCK - 1) / ROWS_PER_BLOCK, 256, 0, stream>>>(
        agg1, wt + 30720, c1_b, h1, N);

    long long total2 = (long long)E * 64;
    conv2_scatter_kernel<<<(unsigned)((total2 + 255) / 256), 256, 0, stream>>>(
        h1, src, dst, dego, degi, agg2, E);

    final_kernel<<<(N + 255) / 256, 256, 0, stream>>>(agg2, c2_w, c2_b, out, N);
}
